// Model_30648886624427
// MI455X (gfx1250) — matrix operand order repaired and checked by reading, not compiled, not run on hardware
//
#include <hip/hip_runtime.h>

typedef __attribute__((ext_vector_type(16))) __bf16 v16bf;
typedef __attribute__((ext_vector_type(8)))  __bf16 v8bf;
typedef __attribute__((ext_vector_type(4)))  __bf16 v4bf;
typedef __attribute__((ext_vector_type(8)))  float  v8f;

#define DIM      768
#define BS       8192
#define BM       64          // rows of nl per block
// ---- slow-path (fused conversion) tile params
#define BN       32
#define KC       128
#define KSTRIDE  136
// ---- fast-path (pre-converted bf16) tile params
#define BN2      128         // cols per block (4 col groups of 32)
#define KC2      64          // K chunk staged in LDS
#define KST2     72          // 64 + 8 pad: 36-dword pitch -> conflict-free b128

__device__ __forceinline__ v16bf ld_frag16(const __bf16* p0, const __bf16* p1) {
  v8bf x = *reinterpret_cast<const v8bf*>(p0);
  v8bf y = *reinterpret_cast<const v8bf*>(p1);
  return __builtin_shufflevector(x, y, 0,1,2,3,4,5,6,7,8,9,10,11,12,13,14,15);
}

#define WMMA_BF16(A, B, C) \
  __builtin_amdgcn_wmma_f32_16x16x32_bf16(false, (A), false, (B), (short)0, (C), false, false)

__device__ __forceinline__ void lse_update(float* m, float* s, const v8f& c) {
#pragma unroll
  for (int v = 0; v < 8; ++v) {
    float x  = c[v];
    float mn = fmaxf(m[v], x);
    s[v] = s[v] * __expf(m[v] - mn) + __expf(x - mn);
    m[v] = mn;
  }
}

// ---------------------------------------------------------------------------
// One-shot f32 -> (bf16 hi, bf16 lo) split of both matrices (memory-bound).
// ---------------------------------------------------------------------------
__global__ __launch_bounds__(256)
void convert_kernel(const float* __restrict__ nl, const float* __restrict__ code,
                    __bf16* __restrict__ Ahg, __bf16* __restrict__ Alg,
                    __bf16* __restrict__ Bhg, __bf16* __restrict__ Blg) {
  size_t base = ((size_t)blockIdx.x * 256 + threadIdx.x) * 4;
  float4 fa = *reinterpret_cast<const float4*>(nl + base);
  float4 fb = *reinterpret_cast<const float4*>(code + base);
  float xa[4] = {fa.x, fa.y, fa.z, fa.w};
  float xb[4] = {fb.x, fb.y, fb.z, fb.w};
  v4bf ah, al, bh, bl;
#pragma unroll
  for (int e = 0; e < 4; ++e) {
    __bf16 h = (__bf16)xa[e]; ah[e] = h; al[e] = (__bf16)(xa[e] - (float)h);
    h = (__bf16)xb[e];        bh[e] = h; bl[e] = (__bf16)(xb[e] - (float)h);
  }
  *reinterpret_cast<v4bf*>(Ahg + base) = ah;
  *reinterpret_cast<v4bf*>(Alg + base) = al;
  *reinterpret_cast<v4bf*>(Bhg + base) = bh;
  *reinterpret_cast<v4bf*>(Blg + base) = bl;
}

// ---------------------------------------------------------------------------
// FAST PATH: 32x32 wave tile (4 C tiles, 16 ds_loads : 12 WMMAs per K-step)
// on pre-split bf16 + online logsumexp.
// ---------------------------------------------------------------------------
__global__ __launch_bounds__(256)
void gemm_lse_fast(const __bf16* __restrict__ Ahg, const __bf16* __restrict__ Alg,
                   const __bf16* __restrict__ Bhg, const __bf16* __restrict__ Blg,
                   float* __restrict__ lse_out) {
  __shared__ __bf16 Ah[BM * KST2],  Al[BM * KST2];    // 64 rows
  __shared__ __bf16 Bh[BN2 * KST2], Bl[BN2 * KST2];   // 128 rows
  __shared__ float  lsePart[4][BM][2];

  const int tid  = threadIdx.x;
  const int wave = tid >> 5;
  const int lane = tid & 31;
  const int half = lane >> 4;
  const int ln   = lane & 15;
  const int rg   = wave & 1;           // 2 row groups of 32 rows
  const int cg   = wave >> 1;          // 4 col groups of 32 cols
  const int blockRow = blockIdx.x * BM;

  float m[2][8], s[2][8];
#pragma unroll
  for (int i = 0; i < 2; ++i)
#pragma unroll
    for (int v = 0; v < 8; ++v) { m[i][v] = -__builtin_inff(); s[i][v] = 0.f; }

  for (int jt = 0; jt < BS / BN2; ++jt) {
    v8f c00 = {}, c01 = {}, c10 = {}, c11 = {};
    for (int kc = 0; kc < DIM / KC2; ++kc) {
      __syncthreads();
      // ---- stage A: 64 rows x 64 bf16 x {hi,lo} = 512 16B-units per array
#pragma unroll
      for (int i = 0; i < 2; ++i) {
        int q  = tid + 256 * i;        // < 512
        int r  = q >> 3;               // 8 v8bf per row
        int k8 = (q & 7) << 3;
        size_t ga = (size_t)(blockRow + r) * DIM + kc * KC2 + k8;
        int la = r * KST2 + k8;
        *reinterpret_cast<v8bf*>(&Ah[la]) = *reinterpret_cast<const v8bf*>(&Ahg[ga]);
        *reinterpret_cast<v8bf*>(&Al[la]) = *reinterpret_cast<const v8bf*>(&Alg[ga]);
      }
      // ---- stage B: 128 rows x 64 bf16 x {hi,lo} = 1024 16B-units per array
#pragma unroll
      for (int i = 0; i < 4; ++i) {
        int q  = tid + 256 * i;        // < 1024
        int r  = q >> 3;
        int k8 = (q & 7) << 3;
        size_t gb = (size_t)(jt * BN2 + r) * DIM + kc * KC2 + k8;
        int la = r * KST2 + k8;
        *reinterpret_cast<v8bf*>(&Bh[la]) = *reinterpret_cast<const v8bf*>(&Bhg[gb]);
        *reinterpret_cast<v8bf*>(&Bl[la]) = *reinterpret_cast<const v8bf*>(&Blg[gb]);
      }
      __syncthreads();
#pragma unroll
      for (int step = 0; step < KC2 / 32; ++step) {
        const int k0 = step * 32;
        const int a0 = (32 * rg + ln) * KST2 + k0 + (half ? 8 : 0);
        const int a1 = (32 * rg + 16 + ln) * KST2 + k0 + (half ? 8 : 0);
        v16bf ah0 = ld_frag16(&Ah[a0], &Ah[a0 + 16]);
        v16bf al0 = ld_frag16(&Al[a0], &Al[a0 + 16]);
        v16bf ah1 = ld_frag16(&Ah[a1], &Ah[a1 + 16]);
        v16bf al1 = ld_frag16(&Al[a1], &Al[a1 + 16]);
        const int b0 = (32 * cg + ln) * KST2 + k0 + half * 8;
        const int b1 = (32 * cg + 16 + ln) * KST2 + k0 + half * 8;
        v16bf bh0 = ld_frag16(&Bh[b0], &Bh[b0 + 16]);
        v16bf bl0 = ld_frag16(&Bl[b0], &Bl[b0 + 16]);
        v16bf bh1 = ld_frag16(&Bh[b1], &Bh[b1 + 16]);
        v16bf bl1 = ld_frag16(&Bl[b1], &Bl[b1 + 16]);
        c00 = WMMA_BF16(ah0, bh0, c00);
        c00 = WMMA_BF16(ah0, bl0, c00);
        c00 = WMMA_BF16(al0, bh0, c00);
        c01 = WMMA_BF16(ah0, bh1, c01);
        c01 = WMMA_BF16(ah0, bl1, c01);
        c01 = WMMA_BF16(al0, bh1, c01);
        c10 = WMMA_BF16(ah1, bh0, c10);
        c10 = WMMA_BF16(ah1, bl0, c10);
        c10 = WMMA_BF16(al1, bh0, c10);
        c11 = WMMA_BF16(ah1, bh1, c11);
        c11 = WMMA_BF16(ah1, bl1, c11);
        c11 = WMMA_BF16(al1, bh1, c11);
      }
    }
    // rows 32*rg+16*i, cols jt*128 + 32*cg + 16*j + ln
    lse_update(m[0], s[0], c00);
    lse_update(m[0], s[0], c01);
    lse_update(m[1], s[1], c10);
    lse_update(m[1], s[1], c11);
  }

#pragma unroll
  for (int off = 8; off >= 1; off >>= 1) {
#pragma unroll
    for (int i = 0; i < 2; ++i)
#pragma unroll
      for (int v = 0; v < 8; ++v) {
        float om = __shfl_xor(m[i][v], off, 32);
        float os = __shfl_xor(s[i][v], off, 32);
        float mn = fmaxf(m[i][v], om);
        s[i][v] = s[i][v] * __expf(m[i][v] - mn) + os * __expf(om - mn);
        m[i][v] = mn;
      }
  }
  if (ln == 0) {
#pragma unroll
    for (int i = 0; i < 2; ++i)
#pragma unroll
      for (int v = 0; v < 8; ++v) {
        int r = 32 * rg + 16 * i + 8 * half + v;
        lsePart[cg][r][0] = m[i][v];
        lsePart[cg][r][1] = s[i][v];
      }
  }
  __syncthreads();
  if (tid < BM) {
    float mn = lsePart[0][tid][0];
#pragma unroll
    for (int g = 1; g < 4; ++g) mn = fmaxf(mn, lsePart[g][tid][0]);
    float ss = 0.f;
#pragma unroll
    for (int g = 0; g < 4; ++g)
      ss += lsePart[g][tid][1] * __expf(lsePart[g][tid][0] - mn);
    lse_out[blockRow + tid] = mn + __logf(ss);
  }
}

// ---------------------------------------------------------------------------
// SLOW PATH (fallback if workspace too small): fused conversion + GEMM+LSE
// ---------------------------------------------------------------------------
__global__ __launch_bounds__(256)
void gemm_lse_kernel(const float* __restrict__ nl,
                     const float* __restrict__ code,
                     float* __restrict__ lse_out) {
  __shared__ __bf16 Ah[BM * KSTRIDE];
  __shared__ __bf16 Al[BM * KSTRIDE];
  __shared__ __bf16 Bh[BN * KSTRIDE];
  __shared__ __bf16 Bl[BN * KSTRIDE];
  __shared__ float  lsePart[2][BM][2];

  const int tid      = threadIdx.x;
  const int wave     = tid >> 5;
  const int lane     = tid & 31;
  const int half     = lane >> 4;
  const int ln       = lane & 15;
  const int rowGroup = wave & 3;
  const int colGroup = wave >> 2;
  const int blockRow = blockIdx.x * BM;

  float m[8], s[8];
#pragma unroll
  for (int v = 0; v < 8; ++v) { m[v] = -__builtin_inff(); s[v] = 0.f; }

  for (int jt = 0; jt < BS / BN; ++jt) {
    v8f c = {};
    for (int kc = 0; kc < DIM / KC; ++kc) {
      __syncthreads();
#pragma unroll
      for (int i = 0; i < 8; ++i) {
        int q  = tid + 256 * i;
        int r  = q >> 5;
        int k4 = (q & 31) << 2;
        const float4 f = *reinterpret_cast<const float4*>(
            nl + (size_t)(blockRow + r) * DIM + kc * KC + k4);
        float xs[4] = {f.x, f.y, f.z, f.w};
        int idx = r * KSTRIDE + k4;
#pragma unroll
        for (int e = 0; e < 4; ++e) {
          __bf16 h = (__bf16)xs[e];
          Ah[idx + e] = h;
          Al[idx + e] = (__bf16)(xs[e] - (float)h);
        }
      }
#pragma unroll
      for (int i = 0; i < 4; ++i) {
        int q  = tid + 256 * i;
        int n  = q >> 5;
        int k4 = (q & 31) << 2;
        const float4 f = *reinterpret_cast<const float4*>(
            code + (size_t)(jt * BN + n) * DIM + kc * KC + k4);
        float xs[4] = {f.x, f.y, f.z, f.w};
        int idx = n * KSTRIDE + k4;
#pragma unroll
        for (int e = 0; e < 4; ++e) {
          __bf16 h = (__bf16)xs[e];
          Bh[idx + e] = h;
          Bl[idx + e] = (__bf16)(xs[e] - (float)h);
        }
      }
      __syncthreads();
#pragma unroll
      for (int step = 0; step < KC / 32; ++step) {
        const int k0 = step * 32;
        const int abase = (16 * rowGroup + ln) * KSTRIDE + k0 + (half ? 8 : 0);
        v16bf a_hi = ld_frag16(&Ah[abase], &Ah[abase + 16]);
        v16bf a_lo = ld_frag16(&Al[abase], &Al[abase + 16]);
        const int bbase = (16 * colGroup + ln) * KSTRIDE + k0 + half * 8;
        v16bf b_hi = ld_frag16(&Bh[bbase], &Bh[bbase + 16]);
        v16bf b_lo = ld_frag16(&Bl[bbase], &Bl[bbase + 16]);
        c = WMMA_BF16(a_hi, b_hi, c);
        c = WMMA_BF16(a_hi, b_lo, c);
        c = WMMA_BF16(a_lo, b_hi, c);
      }
    }
    lse_update(m, s, c);
  }

#pragma unroll
  for (int off = 8; off >= 1; off >>= 1) {
#pragma unroll
    for (int v = 0; v < 8; ++v) {
      float om = __shfl_xor(m[v], off, 32);
      float os = __shfl_xor(s[v], off, 32);
      float mn = fmaxf(m[v], om);
      s[v] = s[v] * __expf(m[v] - mn) + os * __expf(om - mn);
      m[v] = mn;
    }
  }
  if (ln == 0) {
#pragma unroll
    for (int v = 0; v < 8; ++v) {
      int r = 16 * rowGroup + 8 * half + v;
      lsePart[colGroup][r][0] = m[v];
      lsePart[colGroup][r][1] = s[v];
    }
  }
  __syncthreads();
  if (tid < BM) {
    float m0 = lsePart[0][tid][0], s0 = lsePart[0][tid][1];
    float m1 = lsePart[1][tid][0], s1 = lsePart[1][tid][1];
    float mn = fmaxf(m0, m1);
    float ss = s0 * __expf(m0 - mn) + s1 * __expf(m1 - mn);
    lse_out[blockRow + tid] = mn + __logf(ss);
  }
}

// ---------------------------------------------------------------------------
// diag[i] = <nl[i], code[i]>
// ---------------------------------------------------------------------------
__global__ __launch_bounds__(256)
void diag_kernel(const float* __restrict__ nl, const float* __restrict__ code,
                 float* __restrict__ diag_out) {
  int wave = threadIdx.x >> 5, lane = threadIdx.x & 31;
  int row = blockIdx.x * 8 + wave;
  const float* a = nl + (size_t)row * DIM;
  const float* b = code + (size_t)row * DIM;
  float acc = 0.f;
  for (int k = lane; k < DIM; k += 32) acc = fmaf(a[k], b[k], acc);
#pragma unroll
  for (int off = 16; off >= 1; off >>= 1) acc += __shfl_xor(acc, off, 32);
  if (lane == 0) diag_out[row] = acc;
}

// ---------------------------------------------------------------------------
// loss = mean(lse - diag)
// ---------------------------------------------------------------------------
__global__ __launch_bounds__(256)
void loss_kernel(const float* __restrict__ lse, const float* __restrict__ dg,
                 float* __restrict__ out) {
  __shared__ float red[256];
  float acc = 0.f;
  for (int i = threadIdx.x; i < BS; i += 256) acc += lse[i] - dg[i];
  red[threadIdx.x] = acc;
  __syncthreads();
  for (int st = 128; st > 0; st >>= 1) {
    if (threadIdx.x < st) red[threadIdx.x] += red[threadIdx.x + st];
    __syncthreads();
  }
  if (threadIdx.x == 0) out[0] = red[0] / (float)BS;
}

// ---------------------------------------------------------------------------
// out[1..N] = code, out[1+N..2N] = nl  (out+1 only 4B aligned -> b32 copies)
// ---------------------------------------------------------------------------
__global__ __launch_bounds__(256)
void copy_kernel(const float* __restrict__ code, const float* __restrict__ nl,
                 float* __restrict__ out) {
  size_t idx = (size_t)blockIdx.x * 256 + threadIdx.x;
  const size_t N = (size_t)BS * DIM;
  if (idx < N)          out[1 + idx] = code[idx];
  else if (idx < 2 * N) out[1 + idx] = nl[idx - N];
}

extern "C" void kernel_launch(void* const* d_in, const int* in_sizes, int n_in,
                              void* d_out, int out_size, void* d_ws, size_t ws_size,
                              hipStream_t stream) {
  const float* nl   = (const float*)d_in[0];
  const float* code = (const float*)d_in[1];
  float* out = (float*)d_out;
  float* lse = (float*)d_ws;                 // [0, BS)
  float* dg  = lse + BS;                     // [BS, 2*BS)
  const size_t N = (size_t)BS * DIM;

  const size_t need = 2 * BS * sizeof(float) + 4 * N * sizeof(unsigned short);
  if (ws_size >= need) {
    __bf16* Ahg = (__bf16*)(dg + BS);
    __bf16* Alg = Ahg + N;
    __bf16* Bhg = Alg + N;
    __bf16* Blg = Bhg + N;
    convert_kernel<<<(unsigned)(N / 4 / 256), 256, 0, stream>>>(nl, code, Ahg, Alg, Bhg, Blg);
    gemm_lse_fast<<<BS / BM, 256, 0, stream>>>(Ahg, Alg, Bhg, Blg, lse);
  } else {
    gemm_lse_kernel<<<BS / BM, 256, 0, stream>>>(nl, code, lse);
  }
  diag_kernel<<<BS / 8, 256, 0, stream>>>(nl, code, dg);
  loss_kernel<<<1, 256, 0, stream>>>(lse, dg, out);
  copy_kernel<<<(unsigned)((2 * N + 255) / 256), 256, 0, stream>>>(code, nl, out);
}
